// GraphEncoder_57208964383481
// MI455X (gfx1250) — matrix operand order repaired and checked by reading, not compiled, not run on hardware
//
#include <hip/hip_runtime.h>

#define DIM  50
#define DIMK 25
#define ZS   64   // padded row stride for z buffers (zero-padded cols 50..63)

typedef __attribute__((ext_vector_type(16))) _Float16 v16h;
typedef __attribute__((ext_vector_type(8)))  float    v8f;

__device__ __forceinline__ v8f zero8() {
  v8f z;
  for (int i = 0; i < 8; ++i) z[i] = 0.0f;
  return z;
}

__device__ __forceinline__ float lrelu(float x) { return x > 0.0f ? x : 0.2f * x; }

// ---------------- B-fragment pre-packing (cold path, run once per launch) -------------
// Fragment layout per 64x64 virtual B: frag(ks,nt) for lane l = 16 contiguous halves at
// ((ks*4+nt)*32 + l)*16.  B element mapping (contiguous k, as in A): col = nt*16+(l&15),
// K = ks*32 + ((l&16)?16:0) + e.

// Projection B: B[d][f*25+c] = Wtk[f][d][c], zero padded to 64x64.
__global__ void pack_b_proj_kernel(const float* __restrict__ Wtk, _Float16* __restrict__ bp)
{
  const int t = blockIdx.x;                       // node type
  const float* Wt = Wtk + (size_t)t * 2 * DIM * DIMK;
  _Float16* out = bp + (size_t)t * 4096;
  const int tid = threadIdx.x;                    // 256 threads = (ks,nt,lane)
  const int lane = tid & 31;
  const int col  = ((tid >> 5) & 3) * 16 + (lane & 15);
  const int koff = (tid >> 7) * 32 + ((lane & 16) ? 16 : 0);
#pragma unroll
  for (int e = 0; e < 16; ++e) {
    const int k = koff + e;
    float v = 0.0f;
    if (col < 2 * DIMK && k < DIM) {
      const int f = col / DIMK;
      v = Wt[((size_t)f * DIM + k) * DIMK + (col - f * DIMK)];
    }
    out[(size_t)tid * 16 + e] = (_Float16)v;
  }
}

// z-transform B: block-diagonal W, B[f*25+i][f*25+j] = W[i][j], zero padded to 64x64.
__global__ void pack_b_W_kernel(const float* __restrict__ W, _Float16* __restrict__ bp)
{
  const int tid = threadIdx.x;
  const int lane = tid & 31;
  const int col  = ((tid >> 5) & 3) * 16 + (lane & 15);
  const int koff = (tid >> 7) * 32 + ((lane & 16) ? 16 : 0);
#pragma unroll
  for (int e = 0; e < 16; ++e) {
    const int k = koff + e;
    float v = 0.0f;
    if (col < 2 * DIMK && k < 2 * DIMK) {
      const int f = col / DIMK;
      const int i = k - f * DIMK;
      if (i >= 0 && i < DIMK) v = W[(size_t)i * DIMK + (col - f * DIMK)];
    }
    bp[(size_t)tid * 16 + e] = (_Float16)v;
  }
}

__device__ __forceinline__ v16h load_b_frag(const _Float16* __restrict__ bp, int ks, int nt)
{
  const int lane = threadIdx.x & 31;
  return *(const v16h*)(bp + (((size_t)(ks * 4 + nt) * 32 + lane) << 4));
}

// ---------------- Projection: out = l2norm(leaky_relu(emb @ Wtk)) per (factor,node) ---

__global__ void proj_kernel(const float* __restrict__ emb, const _Float16* __restrict__ bp,
                            float* __restrict__ out, int n)
{
  __shared__ float inA[64][68];      // 64 rows staged, zero-padded cols 50..63
  __shared__ float tile[4][16][66];  // epilogue tiles (padded stride)
  const int wave = threadIdx.x >> 5;
  const int lane = threadIdx.x & 31;
  const int row_blk = blockIdx.x * 64;
  const int m0 = row_blk + wave * 16;

  for (int idx = threadIdx.x; idx < 64 * 68; idx += 128) {
    const int r = idx / 68, c = idx - r * 68;
    const int g = row_blk + r;
    float v = 0.0f;
    if (c < DIM && g < n) v = emb[(size_t)g * DIM + c];
    inA[r][c] = v;
  }
  __syncthreads();

  v8f acc[4];
#pragma unroll
  for (int nt = 0; nt < 4; ++nt) acc[nt] = zero8();

  const int lrow  = wave * 16 + (lane & 15);
  const int khalf = (lane & 16) ? 16 : 0;
#pragma unroll
  for (int ks = 0; ks < 2; ++ks) {
    v16h a;
    const float* Ar = inA[lrow];
    const int k0 = ks * 32 + khalf, k1 = k0 + 8;
#pragma unroll
    for (int j = 0; j < 8; ++j) a[j] = (_Float16)Ar[k0 + j];
#pragma unroll
    for (int j = 0; j < 8; ++j) a[8 + j] = (_Float16)Ar[k1 + j];
#pragma unroll
    for (int nt = 0; nt < 4; ++nt) {
      const v16h b = load_b_frag(bp, ks, nt);
      acc[nt] = __builtin_amdgcn_wmma_f32_16x16x32_f16(false, a, false, b, (short)0,
                                                       acc[nt], false, false);
    }
  }

  // D layout: VGPR r -> row = r + 8*(lane>=16), col = nt*16 + (lane&15)
  const int rbase = (lane & 16) ? 8 : 0;
  const int cc = lane & 15;
#pragma unroll
  for (int nt = 0; nt < 4; ++nt)
#pragma unroll
    for (int r = 0; r < 8; ++r)
      tile[wave][rbase + r][nt * 16 + cc] = lrelu(acc[nt][r]);
  __syncthreads();

  const int row = lane & 15, f = (lane >> 4) & 1;
  const int node = m0 + row;
  if (node < n) {
    float vb[DIMK], ss = 0.0f;
#pragma unroll
    for (int j = 0; j < DIMK; ++j) {
      const float v = tile[wave][row][f * DIMK + j];
      vb[j] = v; ss += v * v;
    }
    const float inv = 1.0f / fmaxf(sqrtf(ss), 1e-12f);
#pragma unroll
    for (int j = 0; j < DIMK; ++j)
      out[(size_t)node * DIM + f * DIMK + j] = vb[j] * inv;
  }
}

// ---------------- Edge phase: attention scores, segment softmax, scatter --------------

__global__ void score_kernel(const int* __restrict__ eu, const int* __restrict__ ec,
                             const float* __restrict__ ea, const float* __restrict__ eb,
                             const float* __restrict__ at_e,
                             float* __restrict__ vals, unsigned* __restrict__ rmax, int E)
{
  const int i = blockIdx.x * blockDim.x + threadIdx.x;
  if (i >= E) return;
  const int u = eu[i], c = ec[i];
  const float* pa = ea + (size_t)u * DIM;
  const float* pb = eb + (size_t)c * DIM;
  float s = 0.0f;
#pragma unroll
  for (int k = 0; k < 2; ++k) {
    float d = 0.0f;
#pragma unroll
    for (int j = 0; j < DIMK; ++j) d += at_e[k * DIM + j]        * pa[k * DIMK + j];
#pragma unroll
    for (int j = 0; j < DIMK; ++j) d += at_e[k * DIM + DIMK + j] * pb[k * DIMK + j];
    s += 0.5f * fmaxf(d, 0.0f);   // relu(e_ts_k) * r_n (= 1/FACTOR_K, never updated)
  }
  vals[i] = s;
  atomicMax(rmax + u, __float_as_uint(s));  // s >= 0 => uint order == float order
}

__global__ void expsum_kernel(const int* __restrict__ eu, float* __restrict__ vals,
                              const float* __restrict__ rmax, float* __restrict__ rsum, int E)
{
  const int i = blockIdx.x * blockDim.x + threadIdx.x;
  if (i >= E) return;
  const int u = eu[i];
  const float ex = __expf(vals[i] - rmax[u]);
  vals[i] = ex;
  atomicAdd(rsum + u, ex);
}

__global__ void scatter_kernel(const int* __restrict__ eu, const int* __restrict__ ec,
                               const float* __restrict__ vals, const float* __restrict__ rsum,
                               const float* __restrict__ eb, float* __restrict__ z, int E)
{
  const int i = blockIdx.x * blockDim.x + threadIdx.x;
  if (i >= E) return;
  const int u = eu[i], c = ec[i];
  const float attn = vals[i] / rsum[u];
  const float* pb = eb + (size_t)c * DIM;
  float* pz = z + (size_t)u * ZS;
#pragma unroll
  for (int j = 0; j < DIM; ++j) atomicAdd(pz + j, attn * pb[j]);
}

// ---------------- z-transform: z = leaky_relu(z) @ blockdiag(W); r = softmax_k(tanh(z)·q)

__global__ void transform_kernel(float* z, const _Float16* __restrict__ bp,
                                 const float* __restrict__ q, float* __restrict__ r_out, int n)
{
  __shared__ float tile[4][16][66];
  const int wave = threadIdx.x >> 5;
  const int lane = threadIdx.x & 31;
  const int m0 = (blockIdx.x * 4 + wave) * 16;

  v8f acc[4];
#pragma unroll
  for (int nt = 0; nt < 4; ++nt) acc[nt] = zero8();

  const int arow  = min(m0 + (lane & 15), n - 1);  // clamp: pad rows are discarded
  const int khalf = (lane & 16) ? 16 : 0;
  const float4* Ar = (const float4*)(z + (size_t)arow * ZS);
#pragma unroll
  for (int ks = 0; ks < 2; ++ks) {
    const int i0 = (ks * 32 + khalf) >> 2;        // run of 8 floats (2x float4)
    const int i1 = (ks * 32 + 8 + khalf) >> 2;    // second run of 8 floats
    const float4 a0 = Ar[i0], a1 = Ar[i0 + 1], b0 = Ar[i1], b1 = Ar[i1 + 1];
    v16h a;
    a[0]  = (_Float16)lrelu(a0.x); a[1]  = (_Float16)lrelu(a0.y);
    a[2]  = (_Float16)lrelu(a0.z); a[3]  = (_Float16)lrelu(a0.w);
    a[4]  = (_Float16)lrelu(a1.x); a[5]  = (_Float16)lrelu(a1.y);
    a[6]  = (_Float16)lrelu(a1.z); a[7]  = (_Float16)lrelu(a1.w);
    a[8]  = (_Float16)lrelu(b0.x); a[9]  = (_Float16)lrelu(b0.y);
    a[10] = (_Float16)lrelu(b0.z); a[11] = (_Float16)lrelu(b0.w);
    a[12] = (_Float16)lrelu(b1.x); a[13] = (_Float16)lrelu(b1.y);
    a[14] = (_Float16)lrelu(b1.z); a[15] = (_Float16)lrelu(b1.w);
#pragma unroll
    for (int nt = 0; nt < 4; ++nt) {
      const v16h b = load_b_frag(bp, ks, nt);
      acc[nt] = __builtin_amdgcn_wmma_f32_16x16x32_f16(false, a, false, b, (short)0,
                                                       acc[nt], false, false);
    }
  }

  const int rbase = (lane & 16) ? 8 : 0;
  const int cc = lane & 15;
#pragma unroll
  for (int nt = 0; nt < 4; ++nt)
#pragma unroll
    for (int r = 0; r < 8; ++r)
      tile[wave][rbase + r][nt * 16 + cc] = acc[nt][r];
  __syncthreads();

  const int row = lane & 15, f = (lane >> 4) & 1;
  const int node = m0 + row;
  float t = 0.0f;
  float vb[DIMK];
  if (node < n) {
#pragma unroll
    for (int j = 0; j < DIMK; ++j) {
      const float v = tile[wave][row][f * DIMK + j];
      vb[j] = v;
      t += tanhf(v) * q[j];
    }
  }
  const float to = __shfl_xor(t, 16, 32);     // other factor's logit (wave32)
  if (node < n) {
    const float m = fmaxf(t, to);
    const float e0 = __expf(t - m), e1 = __expf(to - m);
    r_out[(size_t)node * 2 + f] = e0 / (e0 + e1);
#pragma unroll
    for (int j = 0; j < DIMK; ++j)
      z[(size_t)node * ZS + f * DIMK + j] = vb[j];   // overwrite with transformed z
  }
}

// ---------------- Ego update: emb = l2norm(emb + sum_j z_j * r_j), in place ----------

__global__ void ego_user_kernel(float* emb, const float* __restrict__ z,
                                const float* __restrict__ r, int n)
{
  const int idx = blockIdx.x * blockDim.x + threadIdx.x;
  if (idx >= 2 * n) return;
  const int node = idx >> 1, f = idx & 1;
  const size_t be = (size_t)node * DIM + f * DIMK;
  const size_t bz = (size_t)node * ZS  + f * DIMK;
  const float rr = r[(size_t)node * 2 + f];
  float v[DIMK], ss = 0.0f;
#pragma unroll
  for (int j = 0; j < DIMK; ++j) {
    const float x = emb[be + j] + z[bz + j] * rr;
    v[j] = x; ss += x * x;
  }
  const float inv = 1.0f / fmaxf(sqrtf(ss), 1e-12f);
#pragma unroll
  for (int j = 0; j < DIMK; ++j) emb[be + j] = v[j] * inv;
}

__global__ void ego_item_kernel(float* emb, const float* __restrict__ z1,
                                const float* __restrict__ r1,
                                size_t zstride, size_t rstride, int n)
{
  const int idx = blockIdx.x * blockDim.x + threadIdx.x;
  if (idx >= 2 * n) return;
  const int node = idx >> 1, f = idx & 1;
  const size_t be = (size_t)node * DIM + f * DIMK;
  const size_t bz = (size_t)node * ZS  + f * DIMK;
  float v[DIMK], ss = 0.0f;
#pragma unroll
  for (int j = 0; j < DIMK; ++j) v[j] = emb[be + j];
  for (int rel = 0; rel < 7; ++rel) {               // relations 1..7 feed items
    const float* z = z1 + rel * zstride;
    const float rr = r1[rel * rstride + (size_t)node * 2 + f];
#pragma unroll
    for (int j = 0; j < DIMK; ++j) v[j] += z[bz + j] * rr;
  }
#pragma unroll
  for (int j = 0; j < DIMK; ++j) ss += v[j] * v[j];
  const float inv = 1.0f / fmaxf(sqrtf(ss), 1e-12f);
#pragma unroll
  for (int j = 0; j < DIMK; ++j) emb[be + j] = v[j] * inv;
}

// ---------------- Host orchestration (graph-capture safe) -----------------------------

extern "C" void kernel_launch(void* const* d_in, const int* in_sizes, int n_in,
                              void* d_out, int out_size, void* d_ws, size_t ws_size,
                              hipStream_t stream)
{
  (void)n_in; (void)out_size; (void)ws_size;

  const int n0 = in_sizes[8]  / DIM;   // users
  const int n1 = in_sizes[9]  / DIM;   // items
  const int ne = in_sizes[10] / DIM;   // entities per type

  const float* embin[8];
  for (int i = 0; i < 8; ++i) embin[i] = (const float*)d_in[8 + i];
  const float* Wtk = (const float*)d_in[16];   // [8,2,50,25]
  const float* at  = (const float*)d_in[17];   // [8,2,50]
  const float* W   = (const float*)d_in[18];   // [25,25]
  const float* qr  = (const float*)d_in[19];   // [8,25]

  float* out = (float*)d_out;   // [n0 rows | n1 rows] x 50, doubles as live embs 0/1
  float* ws  = (float*)d_ws;

  // workspace layout (floats)
  float* ent  = ws;                                          // 6*ne*50 (fixed ent embs)
  float* zb   = ent + (size_t)6 * ne * DIM;                  // z0..z7, padded stride 64
  float* rb   = zb + ((size_t)n0 + 7 * (size_t)n1) * ZS;     // r0..r7
  float* vals = rb + ((size_t)n0 + 7 * (size_t)n1) * 2;      // per-edge temp
  int maxE = 0;
  for (int e = 0; e < 8; ++e) { const int E = in_sizes[e] / 2; if (E > maxE) maxE = E; }
  float* rmax = vals + maxE;                                 // n0 entries
  float* rsum = rmax + n0;                                   // n0 entries
  size_t off = (size_t)(rsum + n0 - ws);
  off = (off + 15) & ~(size_t)15;                            // 64B align for v16h loads
  _Float16* bpP = (_Float16*)(ws + off);                     // 8 * 4096 packed halves
  _Float16* bpW = bpP + 8 * 4096;                            // 4096 packed halves

  // Pre-pack B operands into WMMA fragment order (cold)
  pack_b_proj_kernel<<<dim3(8), dim3(256), 0, stream>>>(Wtk, bpP);
  pack_b_W_kernel<<<dim3(1), dim3(256), 0, stream>>>(W, bpW);

  const int nsizes[8] = {n0, n1, ne, ne, ne, ne, ne, ne};
  float* proj_out[8];
  proj_out[0] = out;
  proj_out[1] = out + (size_t)n0 * DIM;
  for (int t = 2; t < 8; ++t) proj_out[t] = ent + (size_t)(t - 2) * ne * DIM;

  // Phase 0: factor projection of every node type (WMMA GEMM)
  for (int t = 0; t < 8; ++t) {
    const int n = nsizes[t];
    proj_kernel<<<dim3((n + 63) / 64), dim3(128), 0, stream>>>(
        embin[t], bpP + (size_t)t * 4096, proj_out[t], n);
  }

  // Edge metadata: INDEX = [(0,1),(1,0),(1,2)..(1,7)]
  const int* eu[8]; const int* ec[8]; int Es[8], rows[8];
  const float* aemb[8]; const float* bemb[8];
  float* zp[8]; float* rp[8];
  zp[0] = zb; rp[0] = rb;
  for (int e = 1; e < 8; ++e) {
    zp[e] = zb + (size_t)n0 * ZS + (size_t)(e - 1) * n1 * ZS;
    rp[e] = rb + (size_t)n0 * 2  + (size_t)(e - 1) * n1 * 2;
  }
  for (int e = 0; e < 8; ++e) {
    const int* ed = (const int*)d_in[e];
    Es[e] = in_sizes[e] / 2;
    eu[e] = ed; ec[e] = ed + Es[e];
    rows[e] = (e == 0) ? n0 : n1;
  }
  aemb[0] = out;                      bemb[0] = out + (size_t)n0 * DIM;
  aemb[1] = out + (size_t)n0 * DIM;   bemb[1] = out;
  for (int e = 2; e < 8; ++e) {
    aemb[e] = out + (size_t)n0 * DIM;
    bemb[e] = ent + (size_t)(e - 2) * ne * DIM;
  }

  const size_t zbytes = ((size_t)n0 + 7 * (size_t)n1) * ZS * sizeof(float);

  for (int iter = 0; iter < 4; ++iter) {
    hipMemsetAsync(zb, 0, zbytes, stream);
    for (int e = 0; e < 8; ++e) {
      const int E = Es[e], nr = rows[e];
      hipMemsetAsync(rmax, 0, (size_t)nr * sizeof(float), stream);
      hipMemsetAsync(rsum, 0, (size_t)nr * sizeof(float), stream);
      score_kernel<<<dim3((E + 255) / 256), dim3(256), 0, stream>>>(
          eu[e], ec[e], aemb[e], bemb[e], at + (size_t)e * 2 * DIM,
          vals, (unsigned*)rmax, E);
      expsum_kernel<<<dim3((E + 255) / 256), dim3(256), 0, stream>>>(
          eu[e], vals, rmax, rsum, E);
      scatter_kernel<<<dim3((E + 255) / 256), dim3(256), 0, stream>>>(
          eu[e], ec[e], vals, rsum, bemb[e], zp[e], E);
      transform_kernel<<<dim3((nr + 63) / 64), dim3(128), 0, stream>>>(
          zp[e], bpW, qr + (size_t)e * DIMK, rp[e], nr);
    }
    ego_user_kernel<<<dim3((2 * n0 + 255) / 256), dim3(256), 0, stream>>>(
        out, zp[0], rp[0], n0);
    ego_item_kernel<<<dim3((2 * n1 + 255) / 256), dim3(256), 0, stream>>>(
        out + (size_t)n0 * DIM, zp[1], rp[1],
        (size_t)n1 * ZS, (size_t)n1 * 2, n1);
  }
}
